// Block_32152125178025
// MI455X (gfx1250) — matrix operand order repaired and checked by reading, not compiled, not run on hardware
//
#include <hip/hip_runtime.h>
#include <hip/hip_bf16.h>

// ---------------------------------------------------------------------------
// CDNA5 / gfx1250 WMMA implementation of the detection-pair MLP.
//   f1 = relu(det @ W1 + b1)                          [N,32]   (f16 in ws)
//   comb = [pairFeat | f1[c] | f1[n]]                 [E,96]
//   h = relu(relu(comb@Wp0+bp0) @ Wp1 + bp1)          [E,64]
//   pooled = segment_max(h, 32-pair segments)         [N,64]   (f16 in ws)
//   out = relu(det + relu(relu(pooled@Wq0+bq0)@Wq1+bq1) @ Wo + bo)
// All GEMMs via v_wmma_f32_16x16x32_f16; one wave owns one 32-pair segment.
// ---------------------------------------------------------------------------

typedef __attribute__((ext_vector_type(16))) _Float16 v16h;
typedef __attribute__((ext_vector_type(8)))  _Float16 v8h;
typedef __attribute__((ext_vector_type(8)))  float    v8f;
typedef __attribute__((ext_vector_type(4)))  float    v4f;

#define N_DETS 50000
#define TILES16 3125            // 50000 / 16

__device__ __forceinline__ v8f wmma16(v16h a, v16h b, v8f c) {
  return __builtin_amdgcn_wmma_f32_16x16x32_f16(
      /*neg_a=*/false, a, /*neg_b=*/false, b,
      /*c_mod=*/(short)0, c, /*reuse_a=*/false, /*reuse_b=*/false);
}

// A-fragment (16-bit, 16x32): lane row = lane&15; hi = lane>>4.
// halves 0..7  -> K = 8*hi + {0..7}
// halves 8..15 -> K = 16 + 8*hi + {0..7}
__device__ __forceinline__ v16h load_a_f32(const float* __restrict__ rowp, int hi) {
  const float* p0 = rowp + 8 * hi;
  const float* p1 = rowp + 16 + 8 * hi;
  v4f q0 = *(const v4f*)(p0);
  v4f q1 = *(const v4f*)(p0 + 4);
  v4f q2 = *(const v4f*)(p1);
  v4f q3 = *(const v4f*)(p1 + 4);
  v16h a;
#pragma unroll
  for (int i = 0; i < 4; ++i) {
    a[i]      = (_Float16)q0[i];
    a[4 + i]  = (_Float16)q1[i];
    a[8 + i]  = (_Float16)q2[i];
    a[12 + i] = (_Float16)q3[i];
  }
  return a;
}

__device__ __forceinline__ v16h load_a_f16(const _Float16* __restrict__ rowp, int hi) {
  v8h lo = *(const v8h*)(rowp + 8 * hi);
  v8h hh = *(const v8h*)(rowp + 16 + 8 * hi);
  return __builtin_shufflevector(lo, hh, 0, 1, 2, 3, 4, 5, 6, 7,
                                         8, 9, 10, 11, 12, 13, 14, 15);
}

// Packed B fragment: [frag][lane][16 halves], frag = kt*NT + nt.
__device__ __forceinline__ v16h load_bfrag(const _Float16* __restrict__ wp,
                                           int kt, int nt, int NT, int lane) {
  return *(const v16h*)(wp + (((kt * NT + nt) * 32 + lane) << 4));
}

// C/D 16x16 f32 layout: VGPR r, lane l -> row = r + 8*(l>>4), col = l&15.
// Add bias, relu, store f16 row-major (leading dim ldn halves).
__device__ __forceinline__ void store_cd_relu_f16(_Float16* base, int ldn,
                                                  v8f c, int nt,
                                                  const float* __restrict__ bias,
                                                  int lane) {
  int hi = lane >> 4;
  int col = (lane & 15) + nt * 16;
  float bv = bias[col];
#pragma unroll
  for (int r = 0; r < 8; ++r) {
    float v = c[r] + bv;
    v = v > 0.f ? v : 0.f;
    base[(r + 8 * hi) * ldn + col] = (_Float16)v;
  }
}

// ---------------------------------------------------------------------------
// Kernel 0: pack one f32 weight matrix [K,N] into f16 WMMA B-fragments.
// B layout (16-bit, 32x16): lane col = lane&15; half i -> K = 16*(i>>3)+8*(lane>>4)+(i&7).
// ---------------------------------------------------------------------------
__global__ void pack_b_kernel(const float* __restrict__ W, _Float16* __restrict__ dst,
                              int K, int N) {
  int NT = N >> 4;
  int total = (K >> 5) * NT * 512;
  for (int tid = blockIdx.x * blockDim.x + threadIdx.x; tid < total;
       tid += gridDim.x * blockDim.x) {
    int i    = tid & 15;
    int lane = (tid >> 4) & 31;
    int frag = tid >> 9;
    int nt = frag % NT;
    int kt = frag / NT;
    int k = kt * 32 + ((i >> 3) << 4) + ((lane >> 4) << 3) + (i & 7);
    int n = nt * 16 + (lane & 15);
    dst[tid] = (_Float16)W[k * N + n];
  }
}

// ---------------------------------------------------------------------------
// Kernel 1: f1 = relu(det @ W1 + b1), f16 output.  One wave per 16-row tile.
// ---------------------------------------------------------------------------
__global__ void __launch_bounds__(256)
fc1_kernel(const float* __restrict__ det, const _Float16* __restrict__ W1p,
           const float* __restrict__ b1, _Float16* __restrict__ f1h) {
  int wave = threadIdx.x >> 5;
  int lane = threadIdx.x & 31;
  int tile = blockIdx.x * 8 + wave;
  if (tile >= TILES16) return;               // wave-uniform: EXEC stays all-1s
  int hi = lane >> 4;
  int col = lane & 15;
  const float* rowp = det + (size_t)(tile * 16 + col) * 128;

  v8f c0 = {}, c1 = {};
#pragma unroll
  for (int kt = 0; kt < 4; ++kt) {
    v16h a  = load_a_f32(rowp + kt * 32, hi);
    v16h b0 = load_bfrag(W1p, kt, 0, 2, lane);
    v16h b1f = load_bfrag(W1p, kt, 1, 2, lane);
    c0 = wmma16(a, b0, c0);
    c1 = wmma16(a, b1f, c1);
  }
  _Float16* base = f1h + (size_t)tile * 16 * 32;
  store_cd_relu_f16(base, 32, c0, 0, b1, lane);
  store_cd_relu_f16(base, 32, c1, 1, b1, lane);
}

// ---------------------------------------------------------------------------
// Kernel 2: per-detection pair MLP + segment max.  One wave = one detection
// (exactly 32 pairs = two 16-row M tiles).  40 WMMAs per wave.
// ---------------------------------------------------------------------------
__global__ void __launch_bounds__(256)
pair_kernel(const float* __restrict__ pf, const int* __restrict__ nIdxs,
            const _Float16* __restrict__ f1h,
            const _Float16* __restrict__ Wp0p, const float* __restrict__ bp0,
            const _Float16* __restrict__ Wp1p, const float* __restrict__ bp1,
            _Float16* __restrict__ pooled) {
  __shared__ _Float16 lds[8][32 * 64];       // 32 KB: per-wave 32x64 f16 h-tile
  int wave = threadIdx.x >> 5;
  int lane = threadIdx.x & 31;
  int det  = blockIdx.x * 8 + wave;          // grid = 6250 -> exactly 50000
  int hi = lane >> 4;
  int col = lane & 15;
  _Float16* hl = lds[wave];

  v8f acc[2][4] = {};

  // ---- GEMM1: comb[32x96] @ Wp0[96x64] ----
  {  // kt = 0 : pairFeatures columns 0..31
    v16h a0 = load_a_f32(pf + (size_t)(det * 32 + col) * 32, hi);
    v16h a1 = load_a_f32(pf + (size_t)(det * 32 + 16 + col) * 32, hi);
#pragma unroll
    for (int nt = 0; nt < 4; ++nt) {
      v16h b = load_bfrag(Wp0p, 0, nt, 4, lane);
      acc[0][nt] = wmma16(a0, b, acc[0][nt]);
      acc[1][nt] = wmma16(a1, b, acc[1][nt]);
    }
  }
  {  // kt = 1 : f1[center] (identical for all 32 rows of the segment)
    v16h a = load_a_f16(f1h + (size_t)det * 32, hi);
#pragma unroll
    for (int nt = 0; nt < 4; ++nt) {
      v16h b = load_bfrag(Wp0p, 1, nt, 4, lane);
      acc[0][nt] = wmma16(a, b, acc[0][nt]);
      acc[1][nt] = wmma16(a, b, acc[1][nt]);
    }
  }
  {  // kt = 2 : f1[neighbor] gather (hits L2: f1 is 3.2 MB)
    int n0 = nIdxs[det * 32 + col];
    int n1 = nIdxs[det * 32 + 16 + col];
    v16h a0 = load_a_f16(f1h + (size_t)n0 * 32, hi);
    v16h a1 = load_a_f16(f1h + (size_t)n1 * 32, hi);
#pragma unroll
    for (int nt = 0; nt < 4; ++nt) {
      v16h b = load_bfrag(Wp0p, 2, nt, 4, lane);
      acc[0][nt] = wmma16(a0, b, acc[0][nt]);
      acc[1][nt] = wmma16(a1, b, acc[1][nt]);
    }
  }

  // relu(h1) -> LDS (C/D layout -> row-major f16, leading dim 64)
#pragma unroll
  for (int mt = 0; mt < 2; ++mt)
#pragma unroll
    for (int nt = 0; nt < 4; ++nt)
      store_cd_relu_f16(hl + mt * 16 * 64, 64, acc[mt][nt], nt, bp0, lane);
  asm volatile("s_wait_dscnt 0" ::: "memory");

  // ---- GEMM2: h1[32x64] @ Wp1[64x64] ----
  v8f acc2[2][4] = {};
#pragma unroll
  for (int kt = 0; kt < 2; ++kt) {
    v16h a0 = load_a_f16(hl + (0 * 16 + col) * 64 + kt * 32, hi);
    v16h a1 = load_a_f16(hl + (1 * 16 + col) * 64 + kt * 32, hi);
#pragma unroll
    for (int nt = 0; nt < 4; ++nt) {
      v16h b = load_bfrag(Wp1p, kt, nt, 4, lane);
      acc2[0][nt] = wmma16(a0, b, acc2[0][nt]);
      acc2[1][nt] = wmma16(a1, b, acc2[1][nt]);
    }
  }

  // ---- bias + relu + segment max over all 32 rows ----
#pragma unroll
  for (int nt = 0; nt < 4; ++nt) {
    float bv = bp1[nt * 16 + col];
    float mv = 0.f;                          // relu output is >= 0
#pragma unroll
    for (int r = 0; r < 8; ++r) {
      float v0 = acc2[0][nt][r] + bv;
      float v1 = acc2[1][nt][r] + bv;
      mv = fmaxf(mv, fmaxf(v0, v1));
    }
    mv = fmaxf(mv, __shfl_xor(mv, 16, 32));  // fold lane halves (rows 0-7,16-23 / 8-15,24-31 split)
    if (hi == 0)
      pooled[(size_t)det * 64 + nt * 16 + col] = (_Float16)mv;
  }
}

// ---------------------------------------------------------------------------
// Kernel 3: pooled -> q0 -> q1 -> Wo, residual + relu.  One wave / 16 rows.
// ---------------------------------------------------------------------------
__global__ void __launch_bounds__(256)
post_kernel(const float* __restrict__ det, const _Float16* __restrict__ pooled,
            const _Float16* __restrict__ Wq0p, const float* __restrict__ bq0,
            const _Float16* __restrict__ Wq1p, const float* __restrict__ bq1,
            const _Float16* __restrict__ Wop,  const float* __restrict__ bo,
            float* __restrict__ out) {
  __shared__ _Float16 lds[8][16 * 64];       // 16 KB
  int wave = threadIdx.x >> 5;
  int lane = threadIdx.x & 31;
  int tile = blockIdx.x * 8 + wave;
  if (tile >= TILES16) return;               // wave-uniform
  int hi = lane >> 4;
  int col = lane & 15;
  _Float16* hl = lds[wave];

  // ---- q0: pooled[16x64] @ Wq0[64x64] ----
  v8f acc[4] = {};
#pragma unroll
  for (int kt = 0; kt < 2; ++kt) {
    v16h a = load_a_f16(pooled + (size_t)(tile * 16 + col) * 64 + kt * 32, hi);
#pragma unroll
    for (int nt = 0; nt < 4; ++nt)
      acc[nt] = wmma16(a, load_bfrag(Wq0p, kt, nt, 4, lane), acc[nt]);
  }
#pragma unroll
  for (int nt = 0; nt < 4; ++nt)
    store_cd_relu_f16(hl, 64, acc[nt], nt, bq0, lane);
  asm volatile("s_wait_dscnt 0" ::: "memory");

  // ---- q1: p[16x64] @ Wq1[64x64] ----
  v8f acc2[4] = {};
#pragma unroll
  for (int kt = 0; kt < 2; ++kt) {
    v16h a = load_a_f16(hl + col * 64 + kt * 32, hi);
#pragma unroll
    for (int nt = 0; nt < 4; ++nt)
      acc2[nt] = wmma16(a, load_bfrag(Wq1p, kt, nt, 4, lane), acc2[nt]);
  }
#pragma unroll
  for (int nt = 0; nt < 4; ++nt)
    store_cd_relu_f16(hl, 64, acc2[nt], nt, bq1, lane);
  asm volatile("s_wait_dscnt 0" ::: "memory");

  // ---- output: p[16x64] @ Wo[64x128], residual + relu ----
  v8f acco[8] = {};
#pragma unroll
  for (int kt = 0; kt < 2; ++kt) {
    v16h a = load_a_f16(hl + col * 64 + kt * 32, hi);
#pragma unroll
    for (int nt = 0; nt < 8; ++nt)
      acco[nt] = wmma16(a, load_bfrag(Wop, kt, nt, 8, lane), acco[nt]);
  }
#pragma unroll
  for (int nt = 0; nt < 8; ++nt) {
    int n = nt * 16 + col;
    float bv = bo[n];
#pragma unroll
    for (int r = 0; r < 8; ++r) {
      size_t row = (size_t)tile * 16 + r + 8 * hi;
      float v = det[row * 128 + n] + acco[nt][r] + bv;
      out[row * 128 + n] = v > 0.f ? v : 0.f;
    }
  }
}

// ---------------------------------------------------------------------------
extern "C" void kernel_launch(void* const* d_in, const int* in_sizes, int n_in,
                              void* d_out, int out_size, void* d_ws, size_t ws_size,
                              hipStream_t stream) {
  const float* det  = (const float*)d_in[0];
  (void)d_in[1];                               // cIdxs implicit: c = pair / 32
  const int*   nIdx = (const int*)d_in[2];
  const float* pf   = (const float*)d_in[3];
  const float* W1   = (const float*)d_in[4];
  const float* b1   = (const float*)d_in[5];
  const float* Wp0  = (const float*)d_in[6];
  const float* bp0  = (const float*)d_in[7];
  const float* Wp1  = (const float*)d_in[8];
  const float* bp1  = (const float*)d_in[9];
  const float* Wq0  = (const float*)d_in[10];
  const float* bq0  = (const float*)d_in[11];
  const float* Wq1  = (const float*)d_in[12];
  const float* bq1  = (const float*)d_in[13];
  const float* Wo   = (const float*)d_in[14];
  const float* bo   = (const float*)d_in[15];
  float* out = (float*)d_out;
  (void)in_sizes; (void)n_in; (void)out_size; (void)ws_size;

  char* ws = (char*)d_ws;
  _Float16* f1h    = (_Float16*)(ws);                 // 50000*32*2  = 3.2 MB
  _Float16* pooled = (_Float16*)(ws + 3200000);       // 50000*64*2  = 6.4 MB
  _Float16* W1p    = (_Float16*)(ws + 9600000);       // 4*2*512*2   = 8 KB
  _Float16* Wp0p   = (_Float16*)(ws + 9608192);       // 3*4*512*2   = 12 KB
  _Float16* Wp1p   = (_Float16*)(ws + 9620480);       // 2*4*512*2   = 8 KB
  _Float16* Wq0p   = (_Float16*)(ws + 9628672);       // 8 KB
  _Float16* Wq1p   = (_Float16*)(ws + 9636864);       // 8 KB
  _Float16* Wop    = (_Float16*)(ws + 9645056);       // 2*8*512*2   = 16 KB

  // Pack all weights into f16 WMMA B-fragment layout.
  pack_b_kernel<<<16, 256, 0, stream>>>(W1,  W1p, 128, 32);
  pack_b_kernel<<<24, 256, 0, stream>>>(Wp0, Wp0p, 96, 64);
  pack_b_kernel<<<16, 256, 0, stream>>>(Wp1, Wp1p, 64, 64);
  pack_b_kernel<<<16, 256, 0, stream>>>(Wq0, Wq0p, 64, 64);
  pack_b_kernel<<<16, 256, 0, stream>>>(Wq1, Wq1p, 64, 64);
  pack_b_kernel<<<32, 256, 0, stream>>>(Wo,  Wop, 64, 128);

  fc1_kernel <<<(TILES16 + 7) / 8, 256, 0, stream>>>(det, W1p, b1, f1h);
  pair_kernel<<<N_DETS / 8,        256, 0, stream>>>(pf, nIdx, f1h,
                                                     Wp0p, bp0, Wp1p, bp1, pooled);
  post_kernel<<<(TILES16 + 7) / 8, 256, 0, stream>>>(det, pooled,
                                                     Wq0p, bq0, Wq1p, bq1,
                                                     Wop, bo, out);
}
